// RelationalNetwork_33036888440968
// MI455X (gfx1250) — compile-verified, matrix operand order checked by reading, not run on hardware
//
#include <hip/hip_runtime.h>
#include <hip/hip_bf16.h>

// ---------------- geometry ----------------
#define BB   8
#define DD   16
#define FF   16
#define FH   128
#define LL   256            // D*D
#define MROWS (BB*LL*LL)    // 524288 pair rows
#define ROWS2 (BB*LL)       // 2048 rows for the tail MLP
#define STR  136            // LDS row stride in bf16 (17 * 16B chunks -> conflict free)
#define EPSB 1e-5f

typedef __attribute__((ext_vector_type(16))) __bf16 v16bf;
typedef __attribute__((ext_vector_type(8)))  float  v8f;

// ---------------- helpers ----------------
__device__ __forceinline__ unsigned short f2bf(float f) {
    unsigned u = __float_as_uint(f);
    u += 0x7FFFu + ((u >> 16) & 1u);           // round-to-nearest-even
    return (unsigned short)(u >> 16);
}
__device__ __forceinline__ float bf2f(unsigned short h) {
    return __uint_as_float(((unsigned)h) << 16);
}
__device__ __forceinline__ v16bf load_frag(const unsigned short* p) {
    union { uint4 q[2]; v16bf v; } u;
    u.q[0] = *(const uint4*)(p);
    u.q[1] = *(const uint4*)(p + 8);
    return u.v;
}

// ---------------- init / small kernels ----------------
__global__ void k_init(float* P) {
    int i = blockIdx.x * 256 + threadIdx.x;
    if (i < 1024) P[256 + i] = 0.0f;           // zero all 8 stat arrays (sum/sumsq x4)
    if (i < 128) { P[2304 + i] = 1.0f; P[2432 + i] = 0.0f; }  // ones / zeros
}

__global__ void k_transpose_bf16(const float* __restrict__ W, unsigned short* __restrict__ WT,
                                 int K, int N) {
    int i = blockIdx.x * 256 + threadIdx.x;
    if (i < K * N) { int k = i / N, n = i % N; WT[n * K + k] = f2bf(W[i]); }
}

// hi = [x|coord] @ W1a ; hj = [x|coord] @ W1b   (2048 rows x 128)
__global__ void k_hihj(const float* __restrict__ x, const float* __restrict__ coord,
                       const float* __restrict__ W1a, const float* __restrict__ W1b,
                       float* __restrict__ hi, float* __restrict__ hj) {
    __shared__ float xs[18];
    int b = blockIdx.x >> 8, l = blockIdx.x & 255;
    int tid = threadIdx.x;
    if (tid < 16) xs[tid] = x[b * (FF * LL) + tid * LL + l];
    else if (tid < 18) xs[tid] = coord[l * 2 + (tid - 16)];
    __syncthreads();
    float si = 0.f, sj = 0.f;
#pragma unroll
    for (int c = 0; c < 18; c++) {
        si += xs[c] * W1a[c * FH + tid];
        sj += xs[c] * W1b[c * FH + tid];
    }
    hi[blockIdx.x * FH + tid] = si;
    hj[blockIdx.x * FH + tid] = sj;
}

// Closed-form BN1 stats: var(hi_l + hj_k) = Vi + Vj + cross-term over shared batch
__global__ void k_bn1(const float* __restrict__ hi, const float* __restrict__ hj,
                      const float* __restrict__ g1, const float* __restrict__ be1,
                      float* __restrict__ scaleA, float* __restrict__ shiftA) {
    int c = blockIdx.x;          // 128 blocks, one channel each
    int t = threadIdx.x;         // 256 threads
    __shared__ float Sbi[8], Sbj[8], Ssqi, Ssqj;
    if (t < 8) { Sbi[t] = 0.f; Sbj[t] = 0.f; }
    if (t == 8) { Ssqi = 0.f; }
    if (t == 9) { Ssqj = 0.f; }
    __syncthreads();
    float qi = 0.f, qj = 0.f;
#pragma unroll
    for (int b = 0; b < 8; b++) {
        float vi = hi[(b * LL + t) * FH + c];
        float vj = hj[(b * LL + t) * FH + c];
        atomicAdd(&Sbi[b], vi);
        atomicAdd(&Sbj[b], vj);
        qi += vi * vi; qj += vj * vj;
    }
    atomicAdd(&Ssqi, qi);
    atomicAdd(&Ssqj, qj);
    __syncthreads();
    if (t == 0) {
        float Si = 0.f, Sj = 0.f;
        for (int b = 0; b < 8; b++) { Si += Sbi[b]; Sj += Sbj[b]; }
        float mi = Si / 2048.f, mj = Sj / 2048.f;
        float Vi = Ssqi / 2048.f - mi * mi;
        float Vj = Ssqj / 2048.f - mj * mj;
        float cross = 0.f;
        for (int b = 0; b < 8; b++)
            cross += (Sbi[b] / 256.f - mi) * (Sbj[b] / 256.f - mj);
        cross = 2.f * cross / 8.f;
        float sc = g1[c] * rsqrtf(Vi + Vj + cross + EPSB);
        scaleA[c] = sc;
        shiftA[c] = be1[c] - sc * (mi + mj);    // b1 cancels: mu1 = mi+mj+b1
    }
}

__global__ void k_bn_finalize(const float* __restrict__ sum, const float* __restrict__ sumsq,
                              const float* __restrict__ gamma, const float* __restrict__ beta,
                              float Ninv, float* __restrict__ scale, float* __restrict__ shift) {
    int c = threadIdx.x;
    float m = sum[c] * Ninv;
    float var = sumsq[c] * Ninv - m * m;
    float sc = gamma[c] * rsqrtf(var + EPSB);
    scale[c] = sc;
    shift[c] = beta[c] - sc * m;
}

// ---------------- fused WMMA GEMM (128x128 weights, 64 rows / block) ----------------
// mode 0: A = relu(bn1(hi[l]+hj[k])) built on the fly, out -> Hbuf (bf16) + stats
// mode 1: A = relu(affine(Hbuf)), out -> Hbuf IN PLACE (bf16) + stats
// mode 2: A = opt_relu(affine(Ain f32)), out -> Aout (f32) + stats
__global__ __launch_bounds__(256)
void k_gemm_fused(int mode, int relu_in,
                  const float* __restrict__ hi, const float* __restrict__ hj,
                  unsigned short* Hbuf,
                  const float* __restrict__ Ain, float* __restrict__ Aout,
                  const float* __restrict__ scale, const float* __restrict__ shiftv,
                  const unsigned short* __restrict__ WT,   // bf16 [N=128][K=128]
                  const float* __restrict__ bias,
                  float* __restrict__ sumO, float* __restrict__ sumsqO) {
    __shared__ __align__(16) unsigned short Alds[64 * STR];
    __shared__ __align__(16) unsigned short Wlds[FH * STR];
    __shared__ float sAcc[FH], qAcc[FH];
    int tid = threadIdx.x;
    if (tid < FH) { sAcc[tid] = 0.f; qAcc[tid] = 0.f; }

    // stage weights (bf16, pre-transposed [n][k])
    for (int i = tid; i < FH * FH; i += 256) {
        int n = i >> 7, k = i & 127;
        Wlds[n * STR + k] = WT[i];
    }

    int R0 = blockIdx.x * 64;
    if (mode == 0) {
        int b = blockIdx.x >> 10;
        int kk = (blockIdx.x >> 2) & 255;
        int l0 = (blockIdx.x & 3) * 64;
        for (int i = tid; i < 64 * FH; i += 256) {
            int r = i >> 7, c = i & 127;
            float v = hi[(b * LL + l0 + r) * FH + c] + hj[(b * LL + kk) * FH + c];
            v = fmaxf(scale[c] * v + shiftv[c], 0.f);
            Alds[r * STR + c] = f2bf(v);
        }
    } else if (mode == 1) {
        for (int i = tid; i < 64 * FH; i += 256) {
            int r = i >> 7, c = i & 127;
            float v = bf2f(Hbuf[(R0 + r) * FH + c]);
            v = scale[c] * v + shiftv[c];
            if (relu_in) v = fmaxf(v, 0.f);
            Alds[r * STR + c] = f2bf(v);
        }
    } else {
        for (int i = tid; i < 64 * FH; i += 256) {
            int r = i >> 7, c = i & 127;
            float v = Ain[(R0 + r) * FH + c];
            v = scale[c] * v + shiftv[c];
            if (relu_in) v = fmaxf(v, 0.f);
            Alds[r * STR + c] = f2bf(v);
        }
    }
    __syncthreads();

    int wave = tid >> 5, lane = tid & 31;
    int mt = wave >> 1;                 // 0..3  (16-row M tile)
    int nbase = (wave & 1) * 4;         // 4 N tiles per wave
    int lrow = lane & 15;
    int khalf = (lane >> 4) * 16;       // K half per lane group

    v8f acc[4];
#pragma unroll
    for (int t = 0; t < 4; t++)
#pragma unroll
        for (int e = 0; e < 8; e++) acc[t][e] = 0.f;

#pragma unroll
    for (int ks = 0; ks < 4; ks++) {
        v16bf a = load_frag(&Alds[(mt * 16 + lrow) * STR + ks * 32 + khalf]);
#pragma unroll
        for (int t = 0; t < 4; t++) {
            v16bf bfr = load_frag(&Wlds[((nbase + t) * 16 + lrow) * STR + ks * 32 + khalf]);
            acc[t] = __builtin_amdgcn_wmma_f32_16x16x32_bf16(
                false, a, false, bfr, (short)0, acc[t], false, false);
        }
    }

    // epilogue: bias add, store, per-channel stats.
    // WMMA C-layout: lane j and lane j+16 hold the SAME channel (column), and all
    // 8 accumulator elements of a tile are that channel too -> reduce in registers,
    // fold the lane/lane+16 partner with one cross-lane xor-shuffle, and let only
    // lanes 0..15 issue one ds_add_f32 per stat per tile (64 -> 8 atomics per lane).
#pragma unroll
    for (int t = 0; t < 4; t++) {
        int ch = (nbase + t) * 16 + lrow;
        float bch = bias[ch];
        int rbase = mt * 16 + ((lane < 16) ? 0 : 8);
        float ls = 0.f, lq = 0.f;
#pragma unroll
        for (int e = 0; e < 8; e++) {
            float v = acc[t][e] + bch;
            int grow = R0 + rbase + e;
            if (mode == 2) Aout[grow * FH + ch] = v;
            else           Hbuf[grow * FH + ch] = f2bf(v);
            ls += v;
            lq += v * v;
        }
        ls += __shfl_xor(ls, 16, 32);
        lq += __shfl_xor(lq, 16, 32);
        if (lane < 16) {
            atomicAdd(&sAcc[ch], ls);
            atomicAdd(&qAcc[ch], lq);
        }
    }
    __syncthreads();
    if (tid < FH) {
        atomicAdd(&sumO[tid], sAcc[tid]);
        atomicAdd(&sumsqO[tid], qAcc[tid]);
    }
}

// weighted aggregation over l:  xagg[b,k,c] = sum_l wagg[l]*relu(bn3(H[b,k,l,c])) + bagg
__global__ void k_agg(const unsigned short* __restrict__ H,
                      const float* __restrict__ scale3, const float* __restrict__ shift3,
                      const float* __restrict__ wagg, const float* __restrict__ bagg,
                      float* __restrict__ xagg) {
    __shared__ float wl[LL];
    int tid = threadIdx.x;            // 128 threads = channel
    wl[tid] = wagg[tid];
    wl[tid + 128] = wagg[tid + 128];
    __syncthreads();
    float sc = scale3[tid], sh = shift3[tid];
    int base = blockIdx.x * (LL * FH);
    float s = 0.f;
    for (int l = 0; l < LL; l++) {
        float v = bf2f(H[base + l * FH + tid]);
        s += wl[l] * fmaxf(sc * v + sh, 0.f);
    }
    xagg[blockIdx.x * FH + tid] = s + bagg[0];
}

// out = relu(bn_o2(t2)) @ Wo3 + bo3, transposed to (b, f, d, d)
__global__ void k_out(const float* __restrict__ t2,
                      const float* __restrict__ scaleo2, const float* __restrict__ shifto2,
                      const float* __restrict__ Wo3, const float* __restrict__ bo3,
                      float* __restrict__ out) {
    __shared__ float act[FH];
    int tid = threadIdx.x;
    act[tid] = fmaxf(scaleo2[tid] * t2[blockIdx.x * FH + tid] + shifto2[tid], 0.f);
    __syncthreads();
    if (tid < FF) {
        float s = 0.f;
        for (int k = 0; k < FH; k++) s += act[k] * Wo3[k * FF + tid];
        s += bo3[tid];
        int b = blockIdx.x >> 8, l = blockIdx.x & 255;
        out[(b * FF + tid) * LL + l] = s;
    }
}

// ---------------- launch ----------------
extern "C" void kernel_launch(void* const* d_in, const int* in_sizes, int n_in,
                              void* d_out, int out_size, void* d_ws, size_t ws_size,
                              hipStream_t stream) {
    const float* x     = (const float*)d_in[0];
    const float* W1a   = (const float*)d_in[1];
    const float* W1b   = (const float*)d_in[2];
    // b1 (d_in[3]) cancels out in the folded BN1 affine
    const float* g1g   = (const float*)d_in[4];
    const float* g1b   = (const float*)d_in[5];
    const float* W2    = (const float*)d_in[6];
    const float* b2    = (const float*)d_in[7];
    const float* g2g   = (const float*)d_in[8];
    const float* g2b   = (const float*)d_in[9];
    const float* W3    = (const float*)d_in[10];
    const float* b3    = (const float*)d_in[11];
    const float* g3g   = (const float*)d_in[12];
    const float* g3b   = (const float*)d_in[13];
    const float* Wagg  = (const float*)d_in[14];
    const float* bagg  = (const float*)d_in[15];
    const float* Wf    = (const float*)d_in[16];
    const float* bf    = (const float*)d_in[17];
    const float* fg    = (const float*)d_in[18];
    const float* fb    = (const float*)d_in[19];
    const float* Wo2   = (const float*)d_in[20];
    const float* bo2   = (const float*)d_in[21];
    const float* o2g   = (const float*)d_in[22];
    const float* o2b   = (const float*)d_in[23];
    const float* Wo3   = (const float*)d_in[24];
    const float* bo3   = (const float*)d_in[25];
    const float* coord = (const float*)d_in[26];
    float* out = (float*)d_out;

    // workspace carve-up
    char* ws = (char*)d_ws;
    unsigned short* H   = (unsigned short*)ws;                 // 134,217,728 B (bf16 pair buffer, reused in place)
    size_t off = (size_t)MROWS * FH * 2;
    float* hi   = (float*)(ws + off);  off += (size_t)ROWS2 * FH * 4;
    float* hj   = (float*)(ws + off);  off += (size_t)ROWS2 * FH * 4;
    float* xagg = (float*)(ws + off);  off += (size_t)ROWS2 * FH * 4;
    float* t1   = (float*)(ws + off);  off += (size_t)ROWS2 * FH * 4;
    float* t2   = (float*)(ws + off);  off += (size_t)ROWS2 * FH * 4;
    unsigned short* W2T  = (unsigned short*)(ws + off); off += FH * FH * 2;
    unsigned short* W3T  = (unsigned short*)(ws + off); off += FH * FH * 2;
    unsigned short* WfT  = (unsigned short*)(ws + off); off += FH * FH * 2;
    unsigned short* Wo2T = (unsigned short*)(ws + off); off += FH * FH * 2;
    float* P = (float*)(ws + off);     // parameter/stat block (2560 floats)
    float* scaleA = P + 0,    *shiftA = P + 128;
    float* sum2   = P + 256,  *ssq2   = P + 384;
    float* sum3   = P + 512,  *ssq3   = P + 640;
    float* sumf   = P + 768,  *ssqf   = P + 896;
    float* sumo2  = P + 1024, *ssqo2  = P + 1152;
    float* scale2 = P + 1280, *shift2 = P + 1408;
    float* scale3 = P + 1536, *shift3 = P + 1664;
    float* scalef = P + 1792, *shiftf = P + 1920;
    float* scaleo2= P + 2048, *shifto2= P + 2176;
    float* ones   = P + 2304, *zeros  = P + 2432;

    // 0) zero stat accumulators, set identity affine
    hipLaunchKernelGGL(k_init, dim3(8), dim3(256), 0, stream, P);

    // 1) bf16 transposed weights for WMMA B operands
    hipLaunchKernelGGL(k_transpose_bf16, dim3(64), dim3(256), 0, stream, W2,  W2T,  FH, FH);
    hipLaunchKernelGGL(k_transpose_bf16, dim3(64), dim3(256), 0, stream, W3,  W3T,  FH, FH);
    hipLaunchKernelGGL(k_transpose_bf16, dim3(64), dim3(256), 0, stream, Wf,  WfT,  FH, FH);
    hipLaunchKernelGGL(k_transpose_bf16, dim3(64), dim3(256), 0, stream, Wo2, Wo2T, FH, FH);

    // 2) hi / hj projections and closed-form BN1
    hipLaunchKernelGGL(k_hihj, dim3(ROWS2), dim3(128), 0, stream, x, coord, W1a, W1b, hi, hj);
    hipLaunchKernelGGL(k_bn1, dim3(128), dim3(256), 0, stream, hi, hj, g1g, g1b, scaleA, shiftA);

    // 3) pair GEMM: H = relu(bn1(hi+hj)) @ W2 + b2  (bf16, + bn2 stats)
    hipLaunchKernelGGL(k_gemm_fused, dim3(MROWS / 64), dim3(256), 0, stream,
                       0, 1, hi, hj, H, (const float*)nullptr, (float*)nullptr,
                       scaleA, shiftA, W2T, b2, sum2, ssq2);
    hipLaunchKernelGGL(k_bn_finalize, dim3(1), dim3(128), 0, stream,
                       sum2, ssq2, g2g, g2b, 1.0f / (float)MROWS, scale2, shift2);

    // 4) in-place: H = relu(bn2(H)) @ W3 + b3  (+ bn3 stats)
    hipLaunchKernelGGL(k_gemm_fused, dim3(MROWS / 64), dim3(256), 0, stream,
                       1, 1, (const float*)nullptr, (const float*)nullptr, H,
                       (const float*)nullptr, (float*)nullptr,
                       scale2, shift2, W3T, b3, sum3, ssq3);
    hipLaunchKernelGGL(k_bn_finalize, dim3(1), dim3(128), 0, stream,
                       sum3, ssq3, g3g, g3b, 1.0f / (float)MROWS, scale3, shift3);

    // 5) weighted aggregation over l
    hipLaunchKernelGGL(k_agg, dim3(ROWS2), dim3(128), 0, stream,
                       H, scale3, shift3, Wagg, bagg, xagg);

    // 6) tail MLP (WMMA, fp32 path): t1 = xagg @ Wf + bf
    hipLaunchKernelGGL(k_gemm_fused, dim3(ROWS2 / 64), dim3(256), 0, stream,
                       2, 0, (const float*)nullptr, (const float*)nullptr, (unsigned short*)nullptr,
                       xagg, t1, ones, zeros, WfT, bf, sumf, ssqf);
    hipLaunchKernelGGL(k_bn_finalize, dim3(1), dim3(128), 0, stream,
                       sumf, ssqf, fg, fb, 1.0f / (float)ROWS2, scalef, shiftf);

    // 7) t2 = relu(bn_f(t1)) @ Wo2 + bo2
    hipLaunchKernelGGL(k_gemm_fused, dim3(ROWS2 / 64), dim3(256), 0, stream,
                       2, 1, (const float*)nullptr, (const float*)nullptr, (unsigned short*)nullptr,
                       t1, t2, scalef, shiftf, Wo2T, bo2, sumo2, ssqo2);
    hipLaunchKernelGGL(k_bn_finalize, dim3(1), dim3(128), 0, stream,
                       sumo2, ssqo2, o2g, o2b, 1.0f / (float)ROWS2, scaleo2, shifto2);

    // 8) final projection + output transpose
    hipLaunchKernelGGL(k_out, dim3(ROWS2), dim3(128), 0, stream,
                       t2, scaleo2, shifto2, Wo3, bo3, out);

    (void)in_sizes; (void)n_in; (void)out_size; (void)ws_size;
}
